// TimeSeriesHead_51410758533312
// MI455X (gfx1250) — matrix operand order repaired and checked by reading, not compiled, not run on hardware
//
#include <hip/hip_runtime.h>
#include <hip/hip_bf16.h>
#include <math.h>

#define B_  8
#define S_  4096
#define DI_ 64
#define DM_ 512
#define DH_ 1024   // 2*DM
#define DO_ 64

#define CH_ (B_ * DM_)   // 4096 scan channels
#define NC_ 8            // time chunks
#define LC_ (S_ / NC_)   // 512 steps per chunk

#define DECAY 1.08638239f   // exp(-1) + exp(-0.2)*cos(0.5)
#define FW0 0.1f
#define FW1 0.06155722f
#define FW2 0.04634691f
#define FW3 0.03789291f

typedef __attribute__((ext_vector_type(16))) __bf16        v16bf;
typedef __attribute__((ext_vector_type(8)))  float         v8f;
typedef __attribute__((ext_vector_type(8)))  unsigned int  v8u;

// ---- helpers -------------------------------------------------------------

static __device__ __forceinline__ unsigned int f2bf1(float f) {
  unsigned int u = __builtin_bit_cast(unsigned int, f);
  return (u + 0x7FFFu + ((u >> 16) & 1u)) >> 16;   // round-to-nearest-even
}
static __device__ __forceinline__ unsigned int packbf(float lo, float hi) {
  return f2bf1(lo) | (f2bf1(hi) << 16);
}

static __device__ __forceinline__ v8f wmma_bf16(const unsigned int a[8],
                                                const unsigned int b[8], v8f c) {
  v8u au, bu;
#pragma unroll
  for (int i = 0; i < 8; ++i) { au[i] = a[i]; bu[i] = b[i]; }
  return __builtin_amdgcn_wmma_f32_16x16x32_bf16(
      /*neg_a=*/false, __builtin_bit_cast(v16bf, au),
      /*neg_b=*/false, __builtin_bit_cast(v16bf, bu),
      /*c_mod=*/(short)0, c, /*reuse_a=*/false, /*reuse_b=*/false);
}

// A-fragment K index for 16-bit 16x32 A layout (lane hi = lane>=16)
static __device__ __forceinline__ int a_k(int v, int hi) {
  return ((v & 3) << 1) + ((v >> 2) << 4) + (hi << 3);
}

// ---- kernel 1: pack fp32 (K,N) weight into swizzled bf16 B-fragments -----
__global__ void k_pack(const float* __restrict__ src, unsigned int* __restrict__ dst,
                       int K, int N) {
  int tid  = blockIdx.x * blockDim.x + threadIdx.x;
  int tile = tid >> 5, lane = tid & 31;
  int KT = K >> 5, NT = N >> 4;
  if (tile >= KT * NT) return;
  int nt = tile / KT, kt = tile - nt * KT;
  int n   = (nt << 4) + (lane & 15);
  int khi = (lane >> 4) << 3;
  unsigned int* out = dst + ((size_t)tile * 32 + lane) * 8;
#pragma unroll
  for (int v = 0; v < 8; ++v) {
    int k = (kt << 5) + khi + ((v & 3) << 1) + ((v >> 2) << 4);
    out[v] = packbf(src[(size_t)k * N + n], src[(size_t)(k + 1) * N + n]);
  }
}

// ---- kernel 2: x fp32 -> packed bf16 pairs -------------------------------
__global__ void k_cvt_x(const float* __restrict__ x, unsigned int* __restrict__ xb,
                        int n2) {
  int i = blockIdx.x * blockDim.x + threadIdx.x;
  if (i < n2) {
    float2 v = ((const float2*)x)[i];
    xb[i] = packbf(v.x, v.y);
  }
}

// ---- kernel 3: xp = x @ W_in + b_in  (M=B*S, N=512, K=64) ----------------
__global__ __launch_bounds__(256) void k_input_proj(
    const unsigned int* __restrict__ xb,        // bf16 pairs, (M,64)
    const unsigned int* __restrict__ win_swz,   // KT=2
    const float* __restrict__ b_in,
    float* __restrict__ xp) {
  int wid  = (blockIdx.x * blockDim.x + threadIdx.x) >> 5;
  int lane = threadIdx.x & 31;
  int mt = wid >> 5;            // 32 N-tiles per M-tile
  int nt = wid & 31;
  int l15 = lane & 15, hi = lane >> 4;

  float bias = b_in[(nt << 4) + l15];
  v8f c;
#pragma unroll
  for (int i = 0; i < 8; ++i) c[i] = bias;

  int row = (mt << 4) + l15;
#pragma unroll
  for (int kt = 0; kt < 2; ++kt) {           // K=64 -> 2 steps
    unsigned int a[8], b[8];
#pragma unroll
    for (int v = 0; v < 8; ++v) {
      int k = (kt << 5) + a_k(v, hi);
      a[v] = xb[((size_t)row * DI_ + k) >> 1];
    }
    const unsigned int* bp = win_swz + ((size_t)(nt * 2 + kt) * 32 + lane) * 8;
#pragma unroll
    for (int v = 0; v < 8; ++v) b[v] = bp[v];
    c = wmma_bf16(a, b, c);
  }
#pragma unroll
  for (int v = 0; v < 8; ++v) {
    int m = (mt << 4) + v + (hi << 3);
    xp[(size_t)m * DM_ + (nt << 4) + l15] = c[v];
  }
}

// ---- scan (linear IIR + 4-tap FIR), split in time chunks -----------------
// pass 1: chunk-local final memory value (m_init = 0)
__global__ void k_scan_part(const float* __restrict__ xp, float* __restrict__ Mfin) {
  int g  = blockIdx.x * blockDim.x + threadIdx.x;   // CH_*NC_ threads
  int ch = g & (CH_ - 1);
  int c  = g >> 12;                                  // g / CH_
  int b = ch >> 9, d = ch & (DM_ - 1);
  const float* px = xp + (size_t)b * S_ * DM_ + d;
  int t0 = c * LC_;
  float h1 = 0.f, h2 = 0.f, h3 = 0.f, h4 = 0.f;
  if (c > 0) {
    h1 = px[(size_t)(t0 - 1) * DM_];
    h2 = px[(size_t)(t0 - 2) * DM_];
    h3 = px[(size_t)(t0 - 3) * DM_];
    h4 = px[(size_t)(t0 - 4) * DM_];
  }
  float m = 0.f;
#pragma unroll 4
  for (int t = t0; t < t0 + LC_; ++t) {
    float xt = px[(size_t)t * DM_];
    m = DECAY * m + (FW0 * h1 + FW1 * h2 + FW2 * h3 + FW3 * h4) + xt;
    h4 = h3; h3 = h2; h2 = h1; h1 = xt;
  }
  Mfin[c * CH_ + ch] = m;
}

// pass 2: per-channel prefix over chunk carries: M_c = decay^L * M_{c-1} + F_c
__global__ void k_scan_prefix(const float* __restrict__ Mfin,
                              float* __restrict__ carry, float decayL) {
  int ch = blockIdx.x * blockDim.x + threadIdx.x;    // CH_ threads
  float P = 0.f;
#pragma unroll
  for (int c = 0; c < NC_; ++c) {
    carry[c * CH_ + ch] = P;                         // true m at chunk start - 1
    P = decayL * P + Mfin[c * CH_ + ch];
  }
}

// pass 3: re-run local scan seeded with true carry; comb = bf16(xp + m)
__global__ void k_scan_final(const float* __restrict__ xp,
                             const float* __restrict__ carry,
                             unsigned short* __restrict__ comb) {
  int g  = blockIdx.x * blockDim.x + threadIdx.x;
  int ch = g & (CH_ - 1);
  int c  = g >> 12;
  int b = ch >> 9, d = ch & (DM_ - 1);
  const float* px = xp + (size_t)b * S_ * DM_ + d;
  unsigned short* pc = comb + (size_t)b * S_ * DM_ + d;
  int t0 = c * LC_;
  float h1 = 0.f, h2 = 0.f, h3 = 0.f, h4 = 0.f;
  if (c > 0) {
    h1 = px[(size_t)(t0 - 1) * DM_];
    h2 = px[(size_t)(t0 - 2) * DM_];
    h3 = px[(size_t)(t0 - 3) * DM_];
    h4 = px[(size_t)(t0 - 4) * DM_];
  }
  float m = carry[c * CH_ + ch];
#pragma unroll 4
  for (int t = t0; t < t0 + LC_; ++t) {
    float xt = px[(size_t)t * DM_];
    m = DECAY * m + (FW0 * h1 + FW1 * h2 + FW2 * h3 + FW3 * h4) + xt;
    h4 = h3; h3 = h2; h2 = h1; h1 = xt;
    pc[(size_t)t * DM_] = (unsigned short)f2bf1(xt + m);
  }
}

// ---- kernel 5: fused MLP + residual + output projection ------------------
__global__ __launch_bounds__(256) void k_mlp(
    const unsigned short* __restrict__ comb,   // bf16 (M,512)
    const float* __restrict__ xp,              // fp32 (M,512)
    const unsigned int* __restrict__ w1_swz,   // K=512  KT=16, NT=64
    const float* __restrict__ b1,
    const unsigned int* __restrict__ w2_swz,   // K=1024 KT=32, NT=32
    const float* __restrict__ b2,
    const unsigned int* __restrict__ wout_swz, // K=512  KT=16, NT=4
    const float* __restrict__ b_out,
    float* __restrict__ out) {
  __shared__ unsigned short lds_ao[16 * DM_];  // 16 KB: comb tile, then out tile
  __shared__ unsigned short lds_h[16 * DH_];   // 32 KB gelu(h) tile

  int row0 = blockIdx.x << 4;
  int tid  = threadIdx.x;
  int wave = tid >> 5, lane = tid & 31;
  int l15 = lane & 15, hi = lane >> 4;

  // stage comb tile (16 KB) into LDS with CDNA5 async-to-LDS DMA:
  // 256 threads x 4 x b128, ASYNCcnt-tracked.
  {
    unsigned long long src = (unsigned long long)(uintptr_t)(comb + (size_t)row0 * DM_);
    unsigned int lbase = (unsigned int)(uintptr_t)(&lds_ao[0]);
#pragma unroll
    for (int i = 0; i < 4; ++i) {
      unsigned int off  = (unsigned int)(tid * 16 + i * 4096);
      unsigned int ldsa = lbase + off;
      asm volatile("global_load_async_to_lds_b128 %0, %1, %2 offset:0"
                   :: "v"(ldsa), "v"(off), "s"(src) : "memory");
    }
    asm volatile("s_wait_asynccnt 0x0" ::: "memory");
  }
  __syncthreads();

  // ---- GEMM2: h = gelu(comb @ W1 + b1) ; wave covers N cols [wave*128,+128)
  {
    v8f acc[8];
#pragma unroll
    for (int j = 0; j < 8; ++j) {
      float bias = b1[((wave * 8 + j) << 4) + l15];
#pragma unroll
      for (int i = 0; i < 8; ++i) acc[j][i] = bias;
    }
#pragma unroll 1
    for (int kt = 0; kt < 16; ++kt) {
      unsigned int a[8];
#pragma unroll
      for (int v = 0; v < 8; ++v) {
        int k = (kt << 5) + a_k(v, hi);
        a[v] = *(const unsigned int*)&lds_ao[l15 * DM_ + k];
      }
#pragma unroll
      for (int j = 0; j < 8; ++j) {
        int nt = wave * 8 + j;
        const unsigned int* bp = w1_swz + ((size_t)(nt * 16 + kt) * 32 + lane) * 8;
        unsigned int b[8];
#pragma unroll
        for (int v = 0; v < 8; ++v) b[v] = bp[v];
        acc[j] = wmma_bf16(a, b, acc[j]);
      }
    }
#pragma unroll
    for (int j = 0; j < 8; ++j) {
      int col = ((wave * 8 + j) << 4) + l15;
#pragma unroll
      for (int v = 0; v < 8; ++v) {
        float xg = acc[j][v];
        float g  = 0.5f * xg * (1.0f + erff(xg * 0.70710678118f));   // exact gelu
        int m = v + (hi << 3);
        lds_h[m * DH_ + col] = (unsigned short)f2bf1(g);
      }
    }
  }
  __syncthreads();   // lds_ao fully consumed; safe to overwrite with out tile

  // ---- GEMM3: o = xp + h @ W2 + b2 ; wave covers N cols [wave*64,+64)
  {
    v8f acc[4];
#pragma unroll
    for (int j = 0; j < 4; ++j) {
      float bias = b2[((wave * 4 + j) << 4) + l15];
#pragma unroll
      for (int i = 0; i < 8; ++i) acc[j][i] = bias;
    }
#pragma unroll 1
    for (int kt = 0; kt < 32; ++kt) {
      unsigned int a[8];
#pragma unroll
      for (int v = 0; v < 8; ++v) {
        int k = (kt << 5) + a_k(v, hi);
        a[v] = *(const unsigned int*)&lds_h[l15 * DH_ + k];
      }
#pragma unroll
      for (int j = 0; j < 4; ++j) {
        int nt = wave * 4 + j;
        const unsigned int* bp = w2_swz + ((size_t)(nt * 32 + kt) * 32 + lane) * 8;
        unsigned int b[8];
#pragma unroll
        for (int v = 0; v < 8; ++v) b[v] = bp[v];
        acc[j] = wmma_bf16(a, b, acc[j]);
      }
    }
#pragma unroll
    for (int j = 0; j < 4; ++j) {
      int col = ((wave * 4 + j) << 4) + l15;
#pragma unroll
      for (int v = 0; v < 8; ++v) {
        int m = v + (hi << 3);
        float o = acc[j][v] + xp[(size_t)(row0 + m) * DM_ + col];
        lds_ao[m * DM_ + col] = (unsigned short)f2bf1(o);
      }
    }
  }
  __syncthreads();

  // ---- GEMM4: final = o @ W_out + b_out ; waves 0..3, one 16-col tile each
  if (wave < 4) {
    int nt = wave;
    float bias = b_out[(nt << 4) + l15];
    v8f c;
#pragma unroll
    for (int i = 0; i < 8; ++i) c[i] = bias;
#pragma unroll 1
    for (int kt = 0; kt < 16; ++kt) {
      unsigned int a[8], b[8];
#pragma unroll
      for (int v = 0; v < 8; ++v) {
        int k = (kt << 5) + a_k(v, hi);
        a[v] = *(const unsigned int*)&lds_ao[l15 * DM_ + k];
      }
      const unsigned int* bp = wout_swz + ((size_t)(nt * 16 + kt) * 32 + lane) * 8;
#pragma unroll
      for (int v = 0; v < 8; ++v) b[v] = bp[v];
      c = wmma_bf16(a, b, c);
    }
#pragma unroll
    for (int v = 0; v < 8; ++v) {
      int m = v + (hi << 3);
      out[(size_t)(row0 + m) * DO_ + (nt << 4) + l15] = c[v];
    }
  }
}

// ---- launcher ------------------------------------------------------------
extern "C" void kernel_launch(void* const* d_in, const int* in_sizes, int n_in,
                              void* d_out, int out_size, void* d_ws, size_t ws_size,
                              hipStream_t stream) {
  (void)in_sizes; (void)n_in; (void)out_size; (void)ws_size;
  const float* x     = (const float*)d_in[0];
  const float* W_in  = (const float*)d_in[1];
  const float* b_in  = (const float*)d_in[2];
  const float* W1    = (const float*)d_in[3];
  const float* b1    = (const float*)d_in[4];
  const float* W2    = (const float*)d_in[5];
  const float* b2    = (const float*)d_in[6];
  const float* W_out = (const float*)d_in[7];
  const float* b_out = (const float*)d_in[8];
  float* out = (float*)d_out;

  char* ws = (char*)d_ws;
  size_t off = 0;
  float*          xp       = (float*)(ws + off);          off += (size_t)B_ * S_ * DM_ * 4;
  unsigned short* comb     = (unsigned short*)(ws + off); off += (size_t)B_ * S_ * DM_ * 2;
  unsigned int*   xb       = (unsigned int*)(ws + off);   off += (size_t)B_ * S_ * DI_ * 2;
  unsigned int*   win_swz  = (unsigned int*)(ws + off);   off += (size_t)DI_ * DM_ * 2;
  unsigned int*   w1_swz   = (unsigned int*)(ws + off);   off += (size_t)DM_ * DH_ * 2;
  unsigned int*   w2_swz   = (unsigned int*)(ws + off);   off += (size_t)DH_ * DM_ * 2;
  unsigned int*   wout_swz = (unsigned int*)(ws + off);   off += (size_t)DM_ * DO_ * 2;
  float*          Mfin     = (float*)(ws + off);          off += (size_t)NC_ * CH_ * 4;
  float*          carry    = (float*)(ws + off);          off += (size_t)NC_ * CH_ * 4;

  // 1) repack weights to WMMA B-fragment layout (bf16)
  {
    struct { const float* s; unsigned int* d; int K, N; } jobs[4] = {
      { W_in,  win_swz,  DI_, DM_ }, { W1, w1_swz, DM_, DH_ },
      { W2,    w2_swz,   DH_, DM_ }, { W_out, wout_swz, DM_, DO_ },
    };
    for (int i = 0; i < 4; ++i) {
      int threads = (jobs[i].K / 32) * (jobs[i].N / 16) * 32;
      k_pack<<<(threads + 255) / 256, 256, 0, stream>>>(jobs[i].s, jobs[i].d,
                                                        jobs[i].K, jobs[i].N);
    }
  }
  // 2) x -> bf16
  {
    int n2 = B_ * S_ * DI_ / 2;
    k_cvt_x<<<(n2 + 255) / 256, 256, 0, stream>>>(x, xb, n2);
  }
  // 3) input projection (WMMA)
  k_input_proj<<<(B_ * S_ / 16) * (DM_ / 16) / 8, 256, 0, stream>>>(xb, win_swz, b_in, xp);
  // 4) blocked linear scan -> comb (bf16)
  {
    float decayL = powf(DECAY, (float)LC_);   // decay^512, finite (~2.6e18)
    k_scan_part<<<(CH_ * NC_) / 256, 256, 0, stream>>>(xp, Mfin);
    k_scan_prefix<<<CH_ / 256, 256, 0, stream>>>(Mfin, carry, decayL);
    k_scan_final<<<(CH_ * NC_) / 256, 256, 0, stream>>>(xp, carry, comb);
  }
  // 5) fused MLP + residual + output projection (WMMA)
  k_mlp<<<B_ * S_ / 16, 256, 0, stream>>>(comb, xp, w1_swz, b1, w2_swz, b2,
                                          wout_swz, b_out, out);
}
